// SRlayer__81544249081882
// MI455X (gfx1250) — matrix operand order repaired and checked by reading, not compiled, not run on hardware
//
#include <hip/hip_runtime.h>
#include <hip/hip_bf16.h>

typedef __attribute__((ext_vector_type(16))) _Float16 v16h;
typedef __attribute__((ext_vector_type(8)))  _Float16 v8h;
typedef __attribute__((ext_vector_type(8)))  float    v8f;

#define Bc   8
#define Cc   3
#define Hc   1024
#define Wc   512
#define HWc  (Hc * Wc)            // 524288 = 2^19
#define BHWc (Bc * HWc)           // 4194304
#define NEc  (Bc * Cc * HWc)      // 12582912
#define Dm   512                  // d_model
#define Kt   1536                 // 512 channels * 3 taps
#define Np   (Bc * Hc)            // 8192 output positions

// ---------------------------------------------------------------------------
// Shared-memory radix-2 Cooley-Tukey FFT (DIT, bit-reversed input reorder).
// sign = -1 forward (numpy convention), +1 inverse. scale applied at end.
// ---------------------------------------------------------------------------
__device__ __forceinline__ void fft_shared(float* sre, float* sim, int n, int logn,
                                           int tid, int nt, float sign, float scale) {
    for (int i = tid; i < n; i += nt) {
        int j = (int)(__brev((unsigned)i) >> (32 - logn));
        if (j > i) {
            float tr = sre[i]; sre[i] = sre[j]; sre[j] = tr;
            float ti = sim[i]; sim[i] = sim[j]; sim[j] = ti;
        }
    }
    __syncthreads();
    for (int s = 1; s <= logn; ++s) {
        int half = 1 << (s - 1);
        int m    = 1 << s;
        for (int idx = tid; idx < (n >> 1); idx += nt) {
            int grp = idx >> (s - 1);
            int j   = idx & (half - 1);
            int i0  = (grp << s) + j;
            int i1  = i0 + half;
            float ang = sign * 6.28318530717958647692f * (float)j / (float)m;
            float wi, wr;
            __sincosf(ang, &wi, &wr);
            float xr = sre[i1], xi = sim[i1];
            float tr = wr * xr - wi * xi;
            float ti = wr * xi + wi * xr;
            float ur = sre[i0], ui = sim[i0];
            sre[i1] = ur - tr; sim[i1] = ui - ti;
            sre[i0] = ur + tr; sim[i0] = ui + ti;
        }
        __syncthreads();
    }
    if (scale != 1.0f) {
        for (int i = tid; i < n; i += nt) { sre[i] *= scale; sim[i] *= scale; }
        __syncthreads();
    }
}

// K1: forward FFT along W (real input).  grid = B*C*H blocks of 256.
__global__ void k_fft_w_fwd(const float* __restrict__ x, float* __restrict__ Fre,
                            float* __restrict__ Fim) {
    __shared__ float sre[Wc], sim[Wc];
    size_t base = (size_t)blockIdx.x * Wc;
    int tid = threadIdx.x;
    for (int w = tid; w < Wc; w += 256) { sre[w] = x[base + w]; sim[w] = 0.f; }
    __syncthreads();
    fft_shared(sre, sim, Wc, 9, tid, 256, -1.f, 1.f);
    for (int w = tid; w < Wc; w += 256) { Fre[base + w] = sre[w]; Fim[base + w] = sim[w]; }
}

// K2/K7: FFT along H (in place).  grid = B*C*W blocks of 512.
__global__ void k_fft_h(float* __restrict__ Fre, float* __restrict__ Fim,
                        float sign, float scale) {
    __shared__ float sre[Hc], sim[Hc];
    int bc = blockIdx.x >> 9;
    int w  = blockIdx.x & (Wc - 1);
    int tid = threadIdx.x;
    size_t base = (size_t)bc * HWc + w;
    for (int h = tid; h < Hc; h += 512) {
        size_t off = base + (size_t)h * Wc;
        sre[h] = Fre[off]; sim[h] = Fim[off];
    }
    __syncthreads();
    fft_shared(sre, sim, Hc, 10, tid, 512, sign, scale);
    for (int h = tid; h < Hc; h += 512) {
        size_t off = base + (size_t)h * Wc;
        Fre[off] = sre[h]; Fim[off] = sim[h];
    }
}

// K3/K6: radix-3 DFT along C (in place); optionally emit log-amplitude.
// wi = -sqrt(3)/2 forward, +sqrt(3)/2 inverse; scale = 1 fwd, 1/3 inv.
__global__ void k_dft3(float* __restrict__ Fre, float* __restrict__ Fim,
                       float* __restrict__ LA, float wi, float scale, int writeLA) {
    int idx = blockIdx.x * 256 + threadIdx.x;   // over B*H*W
    if (idx >= BHWc) return;
    int b   = idx >> 19;
    int rem = idx & (HWc - 1);
    size_t p0 = ((size_t)b * 3 + 0) * HWc + rem;
    size_t p1 = p0 + HWc, p2 = p1 + HWc;
    const float wr = -0.5f;
    float a0r = Fre[p0], a0i = Fim[p0];
    float a1r = Fre[p1], a1i = Fim[p1];
    float a2r = Fre[p2], a2i = Fim[p2];
    float x0r = a0r + a1r + a2r;
    float x0i = a0i + a1i + a2i;
    float x1r = a0r + (wr * a1r - wi * a1i) + (wr * a2r + wi * a2i);
    float x1i = a0i + (wr * a1i + wi * a1r) + (wr * a2i - wi * a2r);
    float x2r = a0r + (wr * a1r + wi * a1i) + (wr * a2r - wi * a2i);
    float x2i = a0i + (wr * a1i - wi * a1r) + (wr * a2i + wi * a2r);
    x0r *= scale; x0i *= scale; x1r *= scale; x1i *= scale; x2r *= scale; x2i *= scale;
    Fre[p0] = x0r; Fim[p0] = x0i;
    Fre[p1] = x1r; Fim[p1] = x1i;
    Fre[p2] = x2r; Fim[p2] = x2i;
    if (writeLA) {
        const float eps = 4.539992976e-5f;      // exp(-10)
        LA[p0] = __logf(sqrtf(x0r * x0r + x0i * x0i) + eps);
        LA[p1] = __logf(sqrtf(x1r * x1r + x1i * x1i) + eps);
        LA[p2] = __logf(sqrtf(x2r * x2r + x2i * x2i) + eps);
    }
}

// K4: 3x3 box filter (1/9 weights), summed over C, zero padding.
__global__ void k_avgconv(const float* __restrict__ LA, float* __restrict__ AF) {
    int idx = blockIdx.x * 256 + threadIdx.x;   // over B*H*W
    if (idx >= BHWc) return;
    int b = idx >> 19, rem = idx & (HWc - 1);
    int h = rem >> 9, w = rem & (Wc - 1);
    float s = 0.f;
    for (int c = 0; c < Cc; ++c) {
        size_t pb = ((size_t)b * 3 + c) * HWc;
        for (int dh = -1; dh <= 1; ++dh) {
            int hh = h + dh; if (hh < 0 || hh >= Hc) continue;
            for (int dw = -1; dw <= 1; ++dw) {
                int ww = w + dw; if (ww < 0 || ww >= Wc) continue;
                s += LA[pb + (size_t)hh * Wc + ww];
            }
        }
    }
    AF[idx] = s * (1.f / 9.f);
}

// K5: spectral residual rescale: F <- exp(log_amp - AF) * F / |F| (phase kept).
__global__ void k_spec(float* __restrict__ Fre, float* __restrict__ Fim,
                       const float* __restrict__ LA, const float* __restrict__ AF) {
    int idx = blockIdx.x * 256 + threadIdx.x;   // over NE
    if (idx >= NEc) return;
    int bcs = idx >> 19;
    int b   = bcs / 3;
    int rem = idx & (HWc - 1);
    float re = Fre[idx], im = Fim[idx];
    float r  = sqrtf(re * re + im * im);
    float t  = __expf(LA[idx] - AF[(size_t)b * HWc + rem]);
    if (r > 1e-30f) { float sc = t / r; re *= sc; im *= sc; }
    else            { re = t; im = 0.f; }
    Fre[idx] = re; Fim[idx] = im;
}

// K8: inverse FFT along W + |.|  ->  SR (B,C,H,W).
__global__ void k_ifft_w_abs(const float* __restrict__ Fre, const float* __restrict__ Fim,
                             float* __restrict__ SR) {
    __shared__ float sre[Wc], sim[Wc];
    size_t base = (size_t)blockIdx.x * Wc;
    int tid = threadIdx.x;
    for (int w = tid; w < Wc; w += 256) { sre[w] = Fre[base + w]; sim[w] = Fim[base + w]; }
    __syncthreads();
    fft_shared(sre, sim, Wc, 9, tid, 256, 1.f, 1.f / (float)Wc);
    for (int w = tid; w < Wc; w += 256)
        SR[base + w] = sqrtf(sre[w] * sre[w] + sim[w] * sim[w]);
}

// K9: 3x3 Gaussian summed over C + per-batch sum reduction for mean.
__global__ void k_gauss(const float* __restrict__ SR, float* __restrict__ SRg,
                        float* __restrict__ sums) {
    int b   = blockIdx.y;
    int rem = blockIdx.x * 256 + threadIdx.x;   // over H*W
    int h = rem >> 9, w = rem & (Wc - 1);
    const float gw[3][3] = {{0.0625f, 0.125f, 0.0625f},
                            {0.125f,  0.25f,  0.125f},
                            {0.0625f, 0.125f, 0.0625f}};
    float s = 0.f;
    for (int c = 0; c < Cc; ++c) {
        size_t pb = ((size_t)b * 3 + c) * HWc;
        for (int dh = -1; dh <= 1; ++dh) {
            int hh = h + dh; if (hh < 0 || hh >= Hc) continue;
            for (int dw = -1; dw <= 1; ++dw) {
                int ww = w + dw; if (ww < 0 || ww >= Wc) continue;
                s += gw[dh + 1][dw + 1] * SR[pb + (size_t)hh * Wc + ww];
            }
        }
    }
    SRg[(size_t)b * HWc + rem] = s;
    __shared__ float red[256];
    int tid = threadIdx.x;
    red[tid] = s;
    __syncthreads();
    for (int o = 128; o > 0; o >>= 1) {
        if (tid < o) red[tid] += red[tid + o];
        __syncthreads();
    }
    if (tid == 0) atomicAdd(&sums[b], red[0]);
}

// K10: threshold + gather into fp16 B-matrix, n-major: Bm[n*Kt + (c*3+t)],
//      n = b*1024 + h, value = Y[b, c, (h-1+t) mod 1024] (wrap pad).
__global__ void k_packB(const float* __restrict__ SRg, const float* __restrict__ sums,
                        _Float16* __restrict__ Bm) {
    int idx = blockIdx.x * 256 + threadIdx.x;   // over Np*Kt
    if (idx >= Np * Kt) return;
    int n  = idx / Kt;
    int kk = idx - n * Kt;
    int c  = kk / 3;
    int t  = kk - c * 3;
    int b  = n >> 10;
    int h  = n & (Hc - 1);
    int sh = (h - 1 + t + Hc) & (Hc - 1);
    float v = SRg[((size_t)b * Hc + sh) * Wc + c];
    float m = sums[b] * (1.f / (float)HWc);
    v = (v > m) ? v : 0.f;
    Bm[idx] = (_Float16)v;
}

// K11: weights -> fp16 A-matrix, row-major (d, k=c*3+t) == w_out's own layout.
__global__ void k_packA(const float* __restrict__ w_out, _Float16* __restrict__ Am) {
    int idx = blockIdx.x * 256 + threadIdx.x;   // over Dm*Kt
    if (idx >= Dm * Kt) return;
    Am[idx] = (_Float16)w_out[idx];
}

// K12: GEMM via V_WMMA_F32_16X16X32_F16.  One wave per 16x16 output tile,
// 48 WMMAs along K=1536.  A/B fragment loads use a contiguous k order,
// same for A and B (lane L: m/n = L&15, half g = L>>4; A: K = g*16+e;
// B: K = g*16+e).  C/D: elem r of lane L -> M=r+8g.
__global__ void k_gemm_wmma(const _Float16* __restrict__ Am, const _Float16* __restrict__ Bm,
                            const float* __restrict__ bias, float* __restrict__ out) {
    int wave = (blockIdx.x * blockDim.x + threadIdx.x) >> 5;
    int lane = threadIdx.x & 31;
    const int MT = Dm / 16;                     // 32 tiles along d
    int mt = wave % MT;
    int nt = wave / MT;                         // 512 tiles along positions
    int g  = lane >> 4;
    int m  = mt * 16 + (lane & 15);
    int n  = nt * 16 + (lane & 15);
    const _Float16* Arow = Am + (size_t)m * Kt;
    const _Float16* Brow = Bm + (size_t)n * Kt;
    v8f acc = {};
#pragma unroll 4
    for (int kb = 0; kb < Kt / 32; ++kb) {
        int ka = kb * 32 + g * 16;
        v8h alo = *(const v8h*)(Arow + ka);
        v8h ahi = *(const v8h*)(Arow + ka + 8);
        v16h a;
#pragma unroll
        for (int i = 0; i < 8; ++i) { a[i] = alo[i]; a[i + 8] = ahi[i]; }
        v16h bv = *(const v16h*)(Brow + kb * 32 + g * 16);
        acc = __builtin_amdgcn_wmma_f32_16x16x32_f16(false, a, false, bv,
                                                     (short)0, acc, false, false);
    }
    int d0 = mt * 16 + g * 8;
    int b  = n >> 10;
    int h  = n & (Hc - 1);
    float* op = out + ((size_t)b * Hc + h) * Dm + d0;
#pragma unroll
    for (int r = 0; r < 8; ++r) op[r] = acc[r] + bias[d0 + r];
}

extern "C" void kernel_launch(void* const* d_in, const int* in_sizes, int n_in,
                              void* d_out, int out_size, void* d_ws, size_t ws_size,
                              hipStream_t stream) {
    const float* x     = (const float*)d_in[0];
    const float* w_out = (const float*)d_in[1];
    const float* b_out = (const float*)d_in[2];
    float* out = (float*)d_out;

    float* ws   = (float*)d_ws;
    float* Fre  = ws;                       // NE floats
    float* Fim  = ws + (size_t)NEc;         // NE floats
    float* LA   = ws + 2 * (size_t)NEc;     // NE floats (log-amp, later SR, later packs)
    float* AF   = ws + 3 * (size_t)NEc;     // B*H*W floats (box filter, later SRg)
    float* sums = AF + (size_t)BHWc;        // 8 floats
    // fp16 packs alias the (dead-by-then) SR region:
    _Float16* Bm = (_Float16*)LA;                         // 12.58M halfs (25.2 MB)
    _Float16* Am = (_Float16*)(LA + (size_t)8000000);     // 0.79M halfs, after Bm

    const float S3 = 0.86602540378443864676f;

    // Forward 3-D FFT (W, then H, then C) with log-amplitude.
    k_fft_w_fwd<<<Bc * Cc * Hc, 256, 0, stream>>>(x, Fre, Fim);
    k_fft_h<<<Bc * Cc * Wc, 512, 0, stream>>>(Fre, Fim, -1.f, 1.f);
    k_dft3<<<BHWc / 256, 256, 0, stream>>>(Fre, Fim, LA, -S3, 1.f, 1);

    // Spectral residual.
    k_avgconv<<<BHWc / 256, 256, 0, stream>>>(LA, AF);
    k_spec<<<NEc / 256, 256, 0, stream>>>(Fre, Fim, LA, AF);

    // Inverse 3-D FFT (C, H, W) + magnitude into SR (reuses LA).
    k_dft3<<<BHWc / 256, 256, 0, stream>>>(Fre, Fim, nullptr, S3, 1.f / 3.f, 0);
    k_fft_h<<<Bc * Cc * Wc, 512, 0, stream>>>(Fre, Fim, 1.f, 1.f / (float)Hc);
    k_ifft_w_abs<<<Bc * Cc * Hc, 256, 0, stream>>>(Fre, Fim, LA);

    // Gaussian + per-batch mean.
    hipMemsetAsync(sums, 0, Bc * sizeof(float), stream);
    k_gauss<<<dim3(HWc / 256, Bc), 256, 0, stream>>>(LA, AF, sums);

    // Pack fp16 operands, then WMMA GEMM (+bias, output already (B,H,Dm)).
    k_packB<<<(Np * Kt) / 256, 256, 0, stream>>>(AF, sums, Bm);
    k_packA<<<(Dm * Kt) / 256, 256, 0, stream>>>(w_out, Am);
    k_gemm_wmma<<<(Dm / 16) * (Np / 16) / 8, 256, 0, stream>>>(Am, Bm, b_out, out);
}
